// GIN_37769942401637
// MI455X (gfx1250) — matrix operand order repaired and checked by reading, not compiled, not run on hardware
//
#include <hip/hip_runtime.h>

// ---------------- problem constants (match reference) ----------------
#define N_NODES   100000
#define N_EDGES   1600000
#define DIM_IN    128
#define DIM_HID   256
#define DIM_OUT   64
#define NUM_GRAPHS 512
#define NEG_SLOPE 0.01f
#define L2_EPS    1e-12f

typedef __attribute__((ext_vector_type(16))) _Float16 v16h;
typedef __attribute__((ext_vector_type(8)))  _Float16 v8h;
typedef __attribute__((ext_vector_type(8)))  float    v8f;

// ---------------- utility kernels ----------------
__global__ __launch_bounds__(256) void zero_f32_kernel(float* p, int n) {
    int i = blockIdx.x * 256 + threadIdx.x;
    if (i < n) p[i] = 0.0f;
}

__global__ __launch_bounds__(256) void cvt_f32_to_f16_kernel(const float* __restrict__ in,
                                                             _Float16* __restrict__ out, int n) {
    int i = blockIdx.x * 256 + threadIdx.x;
    if (i < n) out[i] = (_Float16)in[i];
}

// ---------------- edge scatter: agg[dst] += feat[src] ----------------
template<int LOGF>
__global__ __launch_bounds__(256)
void scatter_kernel(const float* __restrict__ feat, const int* __restrict__ src,
                    const int* __restrict__ dst, float* __restrict__ agg, int E) {
    const int F = 1 << LOGF;
    const long long total  = (long long)E << LOGF;
    const long long stride = (long long)gridDim.x * 256;
    for (long long i = (long long)blockIdx.x * 256 + threadIdx.x; i < total; i += stride) {
        int e = (int)(i >> LOGF);
        int f = (int)(i & (F - 1));
        int s = src[e];
        int d = dst[e];
        // hint the next gather row into cache (emits global_prefetch_b8)
        long long nx = i + stride;
        if (nx < total)
            __builtin_prefetch(&feat[((size_t)src[(int)(nx >> LOGF)] << LOGF)], 0, 0);
        atomicAdd(&agg[((size_t)d << LOGF) + f], feat[((size_t)s << LOGF) + f]);
    }
}

// ---------------- WMMA GEMM: Out[M][N] = act( (Aself (+Aagg | *rowscale)) @ W^T + bias ) ----------------
// W is [N][K] row-major f16 (torch Linear layout). Block = 128 threads = 4 waves,
// one 16-row M tile per block, each wave covers NTILES*16 output columns.
template<int K, int NTILES, bool RELU, bool SCALE>
__global__ __launch_bounds__(128)
void gin_gemm_kernel(const float* __restrict__ Aself,
                     const float* __restrict__ Aagg,
                     const float* __restrict__ rowscale,
                     const _Float16* __restrict__ W,
                     const float* __restrict__ bias,
                     float* __restrict__ Out,
                     int M, int N) {
    __shared__ _Float16 As[16 * K];   // 16-row A tile, f16, row-major [16][K]

    const int tid  = threadIdx.x;
    const int row0 = blockIdx.x * 16;

    // Stage A tile: fuse (self + agg) or (self * rowscale), convert f32 -> f16.
    for (int i = tid; i < 16 * K; i += 128) {
        int r = i / K, c = i - r * K;
        int row = row0 + r;
        float v = 0.0f;
        if (row < M) {
            size_t idx = (size_t)row * K + c;
            if (SCALE) v = Aself[idx] * rowscale[row];
            else       v = Aself[idx] + Aagg[idx];
        }
        As[i] = (_Float16)v;
    }
    __syncthreads();

    const int wave  = tid >> 5;      // wave32
    const int lane  = tid & 31;
    const int mlow  = lane & 15;
    const int hi    = lane >> 4;
    const int n0    = wave * (NTILES * 16);

    v8f acc[NTILES] = {};

    #pragma unroll
    for (int kt = 0; kt < K / 32; ++kt) {
        // A fragment (contiguous k order, same as B):
        //   lane (m, hi): halves 0-7  <- K = 16*hi .. 16*hi+7
        //                 halves 8-15 <- K = 16*hi+8 .. 16*hi+15
        v16h afrag;
        {
            const _Float16* ap = &As[mlow * K + kt * 32 + hi * 16];
            v8h lo = *(const v8h*)ap;
            v8h hc = *(const v8h*)(ap + 8);
            #pragma unroll
            for (int i = 0; i < 8; ++i) { afrag[i] = lo[i]; afrag[i + 8] = hc[i]; }
        }
        #pragma unroll
        for (int t = 0; t < NTILES; ++t) {
            // B fragment (32x16): lane (n, hi) holds K = 16*hi .. 16*hi+15 of column n,
            // column n of B == row n of W ([N][K] row-major) -> one contiguous 32B load.
            const _Float16* bp = W + (size_t)(n0 + t * 16 + mlow) * K + kt * 32 + hi * 16;
            v16h bfrag = *(const v16h*)bp;
            acc[t] = __builtin_amdgcn_wmma_f32_16x16x32_f16(
                false, afrag, false, bfrag, (short)0, acc[t], false, false);
        }
    }

    // Epilogue: C/D layout -> lane holds column n0+t*16+(lane&15), VGPR r holds row r+8*(lane>>4).
    #pragma unroll
    for (int t = 0; t < NTILES; ++t) {
        int col = n0 + t * 16 + mlow;
        float bv = bias[col];
        #pragma unroll
        for (int r = 0; r < 8; ++r) {
            int row = row0 + r + 8 * hi;
            if (row < M) {
                float v = acc[t][r] + bv;
                if (RELU) v = (v >= 0.0f) ? v : NEG_SLOPE * v;
                Out[(size_t)row * N + col] = v;
            }
        }
    }
}

// ---------------- pooling ----------------
__global__ __launch_bounds__(256)
void count_kernel(const int* __restrict__ batch, float* __restrict__ counts, int n) {
    int i = blockIdx.x * 256 + threadIdx.x;
    if (i < n) atomicAdd(&counts[batch[i]], 1.0f);
}

__global__ __launch_bounds__(256)
void pool_kernel(const float* __restrict__ h, const int* __restrict__ batch,
                 float* __restrict__ pool, int nodes) {
    int i = blockIdx.x * 256 + threadIdx.x;            // nodes * 256 elements
    if (i < nodes * DIM_HID) {
        int row = i >> 8;                               // DIM_HID == 256
        int f   = i & 255;
        atomicAdd(&pool[((size_t)batch[row] << 8) + f], h[i]);
    }
}

// rowscale[g] = (1/max(cnt,1)) / max(||pool[g]/max(cnt,1)||_2, eps)
__global__ __launch_bounds__(256)
void rowscale_kernel(const float* __restrict__ pool, const float* __restrict__ counts,
                     float* __restrict__ rowscale) {
    __shared__ float red[256];
    int g = blockIdx.x;
    float inv = 1.0f / fmaxf(counts[g], 1.0f);
    float v = pool[(size_t)g * DIM_HID + threadIdx.x] * inv;
    red[threadIdx.x] = v * v;
    __syncthreads();
    for (int s = 128; s > 0; s >>= 1) {
        if ((int)threadIdx.x < s) red[threadIdx.x] += red[threadIdx.x + s];
        __syncthreads();
    }
    if (threadIdx.x == 0) {
        float norm = sqrtf(red[0]);
        rowscale[g] = inv / fmaxf(norm, L2_EPS);
    }
}

// ---------------- host launcher ----------------
extern "C" void kernel_launch(void* const* d_in, const int* in_sizes, int n_in,
                              void* d_out, int out_size, void* d_ws, size_t ws_size,
                              hipStream_t stream) {
    (void)in_sizes; (void)n_in; (void)out_size; (void)ws_size;

    const float* x     = (const float*)d_in[0];
    const int*   eidx  = (const int*)d_in[1];
    const int*   batch = (const int*)d_in[2];
    const float* W1    = (const float*)d_in[3];
    const float* b1    = (const float*)d_in[4];
    const float* W2    = (const float*)d_in[5];
    const float* b2    = (const float*)d_in[6];
    const float* Wf    = (const float*)d_in[7];
    const float* bf    = (const float*)d_in[8];
    const int* src = eidx;
    const int* dst = eidx + N_EDGES;
    float* out = (float*)d_out;

    // ---- carve workspace ----
    char* ws = (char*)d_ws;
    size_t off = 0;
    auto carve = [&](size_t bytes) -> char* {
        char* p = ws + off;
        off = (off + bytes + 255) & ~(size_t)255;
        return p;
    };
    float*    agg1 = (float*)   carve((size_t)N_NODES * DIM_IN  * 4);
    float*    h1   = (float*)   carve((size_t)N_NODES * DIM_HID * 4);
    float*    agg2 = (float*)   carve((size_t)N_NODES * DIM_HID * 4);
    float*    h2   = (float*)   carve((size_t)N_NODES * DIM_HID * 4);
    float*    poolc= (float*)   carve((size_t)(NUM_GRAPHS * DIM_HID + NUM_GRAPHS) * 4); // pool + counts contiguous
    float*    pool = poolc;
    float*    cnts = poolc + (size_t)NUM_GRAPHS * DIM_HID;
    float*    rsc  = (float*)   carve((size_t)NUM_GRAPHS * 4);
    _Float16* W1h  = (_Float16*)carve((size_t)DIM_HID * DIM_IN  * 2);
    _Float16* W2h  = (_Float16*)carve((size_t)DIM_HID * DIM_HID * 2);
    _Float16* Wfh  = (_Float16*)carve((size_t)DIM_OUT * DIM_HID * 2);

    // ---- 0) weights f32 -> f16 ----
    cvt_f32_to_f16_kernel<<<(DIM_HID*DIM_IN  + 255)/256, 256, 0, stream>>>(W1, W1h, DIM_HID*DIM_IN);
    cvt_f32_to_f16_kernel<<<(DIM_HID*DIM_HID + 255)/256, 256, 0, stream>>>(W2, W2h, DIM_HID*DIM_HID);
    cvt_f32_to_f16_kernel<<<(DIM_OUT*DIM_HID + 255)/256, 256, 0, stream>>>(Wf, Wfh, DIM_OUT*DIM_HID);

    // ---- zero accumulators (must happen every call: graph replay reuses ws) ----
    zero_f32_kernel<<<((int)((size_t)N_NODES*DIM_IN)  + 255)/256, 256, 0, stream>>>(agg1, N_NODES*DIM_IN);
    zero_f32_kernel<<<((int)((size_t)N_NODES*DIM_HID) + 255)/256, 256, 0, stream>>>(agg2, N_NODES*DIM_HID);
    zero_f32_kernel<<<(NUM_GRAPHS*DIM_HID + NUM_GRAPHS + 255)/256, 256, 0, stream>>>(poolc, NUM_GRAPHS*DIM_HID + NUM_GRAPHS);

    // ---- 1) conv1: scatter then GEMM ----
    scatter_kernel<7><<<16384, 256, 0, stream>>>(x, src, dst, agg1, N_EDGES);
    gin_gemm_kernel<DIM_IN, 4, true, false><<<N_NODES/16, 128, 0, stream>>>(
        x, agg1, nullptr, W1h, b1, h1, N_NODES, DIM_HID);

    // ---- 2) conv2: scatter then GEMM ----
    scatter_kernel<8><<<16384, 256, 0, stream>>>(h1, src, dst, agg2, N_EDGES);
    gin_gemm_kernel<DIM_HID, 4, true, false><<<N_NODES/16, 128, 0, stream>>>(
        h1, agg2, nullptr, W2h, b2, h2, N_NODES, DIM_HID);

    // ---- 3) global mean pool + L2 normalize scale ----
    count_kernel<<<(N_NODES + 255)/256, 256, 0, stream>>>(batch, cnts, N_NODES);
    pool_kernel <<<(N_NODES*DIM_HID + 255)/256, 256, 0, stream>>>(h2, batch, pool, N_NODES);
    rowscale_kernel<<<NUM_GRAPHS, 256, 0, stream>>>(pool, cnts, rsc);

    // ---- 4) final linear on normalized graph embeddings ----
    gin_gemm_kernel<DIM_HID, 1, false, true><<<NUM_GRAPHS/16, 128, 0, stream>>>(
        pool, pool /*unused*/, rsc, Wfh, bf, out, NUM_GRAPHS, DIM_OUT);
}
